// TopKMoEGate_35184372088963
// MI455X (gfx1250) — matrix operand order repaired and checked by reading, not compiled, not run on hardware
//
#include <hip/hip_runtime.h>
#include <hip/hip_bf16.h>
#include <cstdint>

typedef __attribute__((ext_vector_type(16))) __bf16 v16bf;
typedef __attribute__((ext_vector_type(8)))  float  v8f;

#define D_DIM        2048
#define E_DIM        64
#define NTOK         16384     // 4*4096 tokens
#define ROWS_PER_WG  128       // 8 waves * 16 rows
#define KCHUNK       128
#define ROW_BF16     136       // 128 + 8 pad (elements) -> 272B row stride
#define ROW_BYTES    272
#define WREG         (E_DIM * ROW_BYTES)        // 17408 bytes per (hi|lo) W plane
#define LOGROW_F     68                         // 64 + 4 pad floats
#define LOG_BYTES_PER_WAVE (16 * LOGROW_F * 4)  // 4352 = WREG*2/8 -> exact overlay
#define RES_OFF      (2 * WREG)                 // 34816
#define NW_OFF       (RES_OFF + ROWS_PER_WG * 16) // 36864
#define SMEM_BYTES   (NW_OFF + E_DIM * 4)       // 37120

__global__ __launch_bounds__(256) void moe_gate_wmma_kernel(
    const float* __restrict__ x, const float* __restrict__ W,
    const float* __restrict__ nw, const float* __restrict__ noise,
    float* __restrict__ gated, int* __restrict__ idx_out,
    float* __restrict__ vals_out)
{
    __shared__ __align__(16) unsigned char smem[SMEM_BYTES];

    const int tid   = threadIdx.x;
    const int wave  = tid >> 5;
    const int lane  = tid & 31;
    const int half  = lane >> 4;
    const int l16   = lane & 15;
    const int wgBase = blockIdx.x * ROWS_PER_WG;
    const int rowTok = wgBase + wave * 16 + l16;   // A-matrix row this lane owns

    // cache noise_weight in LDS (covered by the first __syncthreads below)
    if (tid < E_DIM) ((float*)(smem + NW_OFF))[tid] = nw[tid];

    v8f acc[4] = {};  // 4 N-tiles of 16 experts, fp32 accumulators

    const float* xrow = x + (size_t)rowTok * D_DIM;

    for (int chunk = 0; chunk < D_DIM / KCHUNK; ++chunk) {
        __syncthreads();  // previous chunk's B reads done before overwrite
        // ---- stage W K-chunk into LDS as bf16 hi/lo (split-precision) ----
        #pragma unroll
        for (int i = 0; i < 8; ++i) {
            int f4 = tid + i * 256;          // 2048 float4 per chunk
            int e  = f4 >> 5;                // 32 float4 per expert row
            int kk = (f4 & 31) * 4;
            float4 w4 = *(const float4*)(W + (size_t)e * D_DIM + chunk * KCHUNK + kk);
            __bf16* hi = (__bf16*)(smem)        + e * ROW_BF16 + kk;
            __bf16* lo = (__bf16*)(smem + WREG) + e * ROW_BF16 + kk;
            float wf[4] = {w4.x, w4.y, w4.z, w4.w};
            #pragma unroll
            for (int j = 0; j < 4; ++j) {
                __bf16 h = (__bf16)wf[j];
                hi[j] = h;
                lo[j] = (__bf16)(wf[j] - (float)h);
            }
        }
        if (chunk + 1 < D_DIM / KCHUNK)      // gfx1250 global_prefetch_b8
            __builtin_prefetch(xrow + (chunk + 1) * KCHUNK, 0, 3);
        __syncthreads();

        #pragma unroll
        for (int kt = 0; kt < KCHUNK / 32; ++kt) {
            const int kb = chunk * KCHUNK + kt * 32;
            // ---- A fragment: lane-half h holds K in [16h,16h+16) ----
            union { v16bf v; __bf16 e[16]; } ahi, alo;
            {
                const float* p = xrow + kb + half * 16;
                float4 a0 = *(const float4*)(p);
                float4 a1 = *(const float4*)(p + 4);
                float4 a2 = *(const float4*)(p + 8);
                float4 a3 = *(const float4*)(p + 12);
                float av[16] = {a0.x,a0.y,a0.z,a0.w, a1.x,a1.y,a1.z,a1.w,
                                a2.x,a2.y,a2.z,a2.w, a3.x,a3.y,a3.z,a3.w};
                #pragma unroll
                for (int j = 0; j < 16; ++j) {
                    __bf16 h = (__bf16)av[j];
                    ahi.e[j] = h;
                    alo.e[j] = (__bf16)(av[j] - (float)h);
                }
            }
            // ---- 4 N-tiles: B fragment = contiguous 16-K run per lane-half ----
            #pragma unroll
            for (int t = 0; t < 4; ++t) {
                union { v16bf v; uint4 q[2]; } bhi, blo;
                const int e = t * 16 + l16;
                const unsigned char* bh =
                    smem + e * ROW_BYTES + (kt * 32 + half * 16) * 2;
                bhi.q[0] = *(const uint4*)(bh);
                bhi.q[1] = *(const uint4*)(bh + 16);
                blo.q[0] = *(const uint4*)(bh + WREG);
                blo.q[1] = *(const uint4*)(bh + WREG + 16);
                v8f a = acc[t];
                // logits = hi*hi + hi*lo + lo*hi  (~fp32 accuracy)
                a = __builtin_amdgcn_wmma_f32_16x16x32_bf16(
                        false, ahi.v, false, bhi.v, (short)0, a, false, false);
                a = __builtin_amdgcn_wmma_f32_16x16x32_bf16(
                        false, ahi.v, false, blo.v, (short)0, a, false, false);
                a = __builtin_amdgcn_wmma_f32_16x16x32_bf16(
                        false, alo.v, false, bhi.v, (short)0, a, false, false);
                acc[t] = a;
            }
        }
    }

    // ---- epilogue: dump C tiles to LDS (overlays W stage after barrier) ----
    __syncthreads();
    {
        float* base = (float*)(smem + wave * LOG_BYTES_PER_WAVE);
        #pragma unroll
        for (int t = 0; t < 4; ++t)
            #pragma unroll
            for (int v = 0; v < 8; ++v) {
                int m = v + 8 * half;         // C layout: lane=n, vgpr=m
                int n = t * 16 + l16;
                base[m * LOGROW_F + n] = acc[t][v];
            }
    }
    __syncthreads();

    // ---- per-token noisy top-2 + 2-way softmax (lanes 0-15, one row each) ----
    if (lane < 16) {
        const int r   = lane;
        const int tok = wgBase + wave * 16 + r;
        const float* lgrow = (const float*)(smem + wave * LOG_BYTES_PER_WAVE) + r * LOGROW_F;
        const float* nzrow = noise + (size_t)tok * E_DIM;
        const float* nwl   = (const float*)(smem + NW_OFF);
        float v1 = -__builtin_inff(), v2 = -__builtin_inff();
        int   i1 = 0, i2 = 0;
        #pragma unroll 4
        for (int e4 = 0; e4 < 16; ++e4) {
            float4 lg  = *(const float4*)(lgrow + e4 * 4);
            float4 nz  = *(const float4*)(nzrow + e4 * 4);
            float4 nwv = *(const float4*)(nwl   + e4 * 4);
            float c[4] = {lg.x + nz.x * nwv.x, lg.y + nz.y * nwv.y,
                          lg.z + nz.z * nwv.z, lg.w + nz.w * nwv.w};
            #pragma unroll
            for (int j = 0; j < 4; ++j) {
                int e = e4 * 4 + j;
                if (c[j] > v1)      { v2 = v1; i2 = i1; v1 = c[j]; i1 = e; }
                else if (c[j] > v2) { v2 = c[j]; i2 = e; }
            }
        }
        float ex  = __expf(v2 - v1);          // v2 <= v1, stable
        float inv = 1.0f / (1.0f + ex);
        const int lrow = wave * 16 + r;
        uint4 rr;
        rr.x = (unsigned)i1; rr.y = (unsigned)i2;
        rr.z = __float_as_uint(inv); rr.w = __float_as_uint(ex * inv);
        *(uint4*)(smem + RES_OFF + lrow * 16) = rr;
        ((int2*)idx_out)[tok]   = make_int2(i1, i2);
        ((float2*)vals_out)[tok] = make_float2(v1, v2);
    }
    __syncthreads();

    // ---- coalesced scatter of sparse softmax probs ----
    for (int f = tid; f < ROWS_PER_WG * E_DIM; f += 256) {
        int lrow = f >> 6, e = f & 63;
        uint4 rr = *(const uint4*)(smem + RES_OFF + lrow * 16);
        float val = (e == (int)rr.x) ? __uint_as_float(rr.z)
                  : (e == (int)rr.y) ? __uint_as_float(rr.w) : 0.0f;
        gated[(size_t)(wgBase + lrow) * E_DIM + e] = val;
    }
}

extern "C" void kernel_launch(void* const* d_in, const int* in_sizes, int n_in,
                              void* d_out, int out_size, void* d_ws, size_t ws_size,
                              hipStream_t stream) {
    const float* x     = (const float*)d_in[0];
    const float* W     = (const float*)d_in[1];
    const float* nw    = (const float*)d_in[2];
    const float* noise = (const float*)d_in[3];
    float* gated    = (float*)d_out;                          // [16384*64]
    int*   idx_out  = (int*)(gated + (size_t)NTOK * E_DIM);   // [16384*2] int32 bits
    float* vals_out = (float*)(idx_out + NTOK * 2);           // [16384*2]
    moe_gate_wmma_kernel<<<NTOK / ROWS_PER_WG, 256, 0, stream>>>(
        x, W, nw, noise, gated, idx_out, vals_out);
}
